// MoE_80221399155330
// MI455X (gfx1250) — matrix operand order repaired and checked by reading, not compiled, not run on hardware
//
#include <hip/hip_runtime.h>
#include <math.h>

// ---------------------------------------------------------------------------
// MoE dense-expert forward for MI455X (gfx1250, wave32, WMMA bf16 path)
// ---------------------------------------------------------------------------
static constexpr int Bn = 8192;
static constexpr int Dd = 768;
static constexpr int Gg = 384;
static constexpr int Ee = 8;
static constexpr int Hh = 3072;

typedef __attribute__((ext_vector_type(16))) __bf16          v16bf;
typedef __attribute__((ext_vector_type(8)))  float           v8f;
typedef __attribute__((ext_vector_type(8)))  unsigned short  v8u16;
typedef __attribute__((ext_vector_type(4)))  int             v4i;

typedef __attribute__((address_space(1))) v4i* gv4i_p;   // global
typedef __attribute__((address_space(3))) v4i* lv4i_p;   // LDS

union Frag16 {
    v16bf v;
    v8u16 h[2];
};

#if defined(__HIP_DEVICE_COMPILE__) && __has_builtin(__builtin_amdgcn_global_load_async_to_lds_b128)
#define HAVE_ASYNC_LDS 1
#else
#define HAVE_ASYNC_LDS 0
#endif

__device__ __forceinline__ unsigned short f2bfu(float f) {
    unsigned u = __float_as_uint(f);
    return (unsigned short)((u + 0x7FFFu + ((u >> 16) & 1u)) >> 16); // RNE
}
__device__ __forceinline__ float bf2f(unsigned short s) {
    return __uint_as_float(((unsigned)s) << 16);
}
__device__ __forceinline__ float gelu_exact(float x) {
    return 0.5f * x * (1.0f + erff(x * 0.70710678118654752440f));
}

// ---------------------------------------------------------------------------
// Weight conversion + transpose kernels (fp32 -> bf16 bits)
// ---------------------------------------------------------------------------
__global__ __launch_bounds__(256) void cvt_gw1_kernel(const float* __restrict__ in,
                                                      unsigned short* __restrict__ out) {
    int idx = blockIdx.x * 256 + threadIdx.x;          // [Gg][Dd]
    if (idx >= Gg * Dd) return;
    int n = idx / Dd, d = idx % Dd;
    out[idx] = f2bfu(in[d * Gg + n]);                  // gw1 is [D][Gh]
}

__global__ __launch_bounds__(256) void cvt_ew1_kernel(const float* __restrict__ in,
                                                      unsigned short* __restrict__ out) {
    size_t idx = (size_t)blockIdx.x * 256 + threadIdx.x;   // [E][H][D]
    if (idx >= (size_t)Ee * Hh * Dd) return;
    int d = (int)(idx % Dd);
    size_t r = idx / Dd;
    int hc = (int)(r % Hh);
    int e  = (int)(r / Hh);
    out[idx] = f2bfu(in[((size_t)e * Dd + d) * Hh + hc]);  // ew1 is [E][D][H]
}

__global__ __launch_bounds__(256) void cvt_ew2_kernel(const float* __restrict__ in,
                                                      unsigned short* __restrict__ out) {
    size_t idx = (size_t)blockIdx.x * 256 + threadIdx.x;   // [E][D][H]
    if (idx >= (size_t)Ee * Dd * Hh) return;
    int hr = (int)(idx % Hh);
    size_t r = idx / Hh;
    int dc = (int)(r % Dd);
    int e  = (int)(r / Dd);
    out[idx] = f2bfu(in[((size_t)e * Hh + hr) * Dd + dc]); // ew2 is [E][H][D]
}

// ---------------------------------------------------------------------------
// LayerNorm: one 256-thread block per row of 768
// ---------------------------------------------------------------------------
__global__ __launch_bounds__(256) void ln_kernel(const float* __restrict__ x,
                                                 const float* __restrict__ gamma,
                                                 const float* __restrict__ beta,
                                                 unsigned short* __restrict__ h) {
    __shared__ float s1[256];
    __shared__ float s2[256];
    const int row = blockIdx.x;
    const int tid = threadIdx.x;
    const float* xr = x + (size_t)row * Dd;
    float v0 = xr[tid], v1 = xr[tid + 256], v2 = xr[tid + 512];
    s1[tid] = v0 + v1 + v2;
    s2[tid] = v0 * v0 + v1 * v1 + v2 * v2;
    __syncthreads();
    for (int off = 128; off > 0; off >>= 1) {
        if (tid < off) { s1[tid] += s1[tid + off]; s2[tid] += s2[tid + off]; }
        __syncthreads();
    }
    const float mu  = s1[0] * (1.0f / Dd);
    const float var = s2[0] * (1.0f / Dd) - mu * mu;
    const float rs  = rsqrtf(var + 1e-5f);
    unsigned short* hr = h + (size_t)row * Dd;
    hr[tid]       = f2bfu((v0 - mu) * rs * gamma[tid]       + beta[tid]);
    hr[tid + 256] = f2bfu((v1 - mu) * rs * gamma[tid + 256] + beta[tid + 256]);
    hr[tid + 512] = f2bfu((v2 - mu) * rs * gamma[tid + 512] + beta[tid + 512]);
}

// ---------------------------------------------------------------------------
// Gating layer 1: g = gelu(h @ gw1 + gb1), one wave per 16x16 output tile
// ---------------------------------------------------------------------------
__global__ __launch_bounds__(32) void gate1_kernel(const unsigned short* __restrict__ h,
                                                   const unsigned short* __restrict__ w1t, // [Gg][Dd]
                                                   const float* __restrict__ b1,
                                                   unsigned short* __restrict__ g) {
    const int row0 = blockIdx.x * 16;
    const int n0   = blockIdx.y * 16;
    const int lane = threadIdx.x;
    const int lh = lane & 15;
    const int hi = lane >> 4;

    v8f acc = {0.f, 0.f, 0.f, 0.f, 0.f, 0.f, 0.f, 0.f};
    const unsigned short* arow = h   + (size_t)(row0 + lh) * Dd;
    const unsigned short* brow = w1t + (size_t)(n0  + lh) * Dd;
    for (int k = 0; k < Dd; k += 32) {
        Frag16 a, b;
        a.h[0] = *(const v8u16*)&arow[k + hi * 8];
        a.h[1] = *(const v8u16*)&arow[k + 16 + hi * 8];
        b.h[0] = *(const v8u16*)&brow[k + hi * 8];
        b.h[1] = *(const v8u16*)&brow[k + 16 + hi * 8];
        acc = __builtin_amdgcn_wmma_f32_16x16x32_bf16(false, a.v, false, b.v,
                                                      (short)0, acc, false, false);
    }
    const float bias = b1[n0 + lh];
#pragma unroll
    for (int i = 0; i < 8; ++i) {
        g[(size_t)(row0 + i + hi * 8) * Gg + n0 + lh] = f2bfu(gelu_exact(acc[i] + bias));
    }
}

// ---------------------------------------------------------------------------
// Gating layer 2 + softmax + top-2 (thread per row; negligible FLOPs)
// cw per row: {w0, w1, bits(i0), bits(i1)}
// ---------------------------------------------------------------------------
__global__ __launch_bounds__(256) void gate2_kernel(const unsigned short* __restrict__ g,
                                                    const float* __restrict__ w2, // [Gg][Ee]
                                                    const float* __restrict__ b2,
                                                    float* __restrict__ gate_out,
                                                    float* __restrict__ cw) {
    const int row = blockIdx.x * 256 + threadIdx.x;
    if (row >= Bn) return;
    float lg[Ee];
#pragma unroll
    for (int e = 0; e < Ee; ++e) lg[e] = b2[e];
    const unsigned short* gr = g + (size_t)row * Gg;
    for (int k = 0; k < Gg; ++k) {
        const float gv = bf2f(gr[k]);
#pragma unroll
        for (int e = 0; e < Ee; ++e) lg[e] += gv * w2[k * Ee + e];
    }
    float m = lg[0];
#pragma unroll
    for (int e = 1; e < Ee; ++e) m = fmaxf(m, lg[e]);
    float p[Ee], sum = 0.f;
#pragma unroll
    for (int e = 0; e < Ee; ++e) { p[e] = expf(lg[e] - m); sum += p[e]; }
    const float inv = 1.0f / sum;
    float* go = gate_out + (size_t)row * Ee;
#pragma unroll
    for (int e = 0; e < Ee; ++e) { p[e] *= inv; go[e] = p[e]; }
    int i0 = 0;
#pragma unroll
    for (int e = 1; e < Ee; ++e) if (p[e] > p[i0]) i0 = e;
    int i1 = (i0 == 0) ? 1 : 0;
#pragma unroll
    for (int e = 0; e < Ee; ++e) if (e != i0 && p[e] > p[i1]) i1 = e;
    const float t = p[i0] + p[i1];
    float* c = cw + (size_t)row * 4;
    c[0] = p[i0] / t;
    c[1] = p[i1] / t;
    c[2] = __int_as_float(i0);
    c[3] = __int_as_float(i1);
}

// ---------------------------------------------------------------------------
// Fused expert FFN: per block = (32-row tile, one expert), 8 waves.
//   wave w: rh = w>>2 selects the 16-row half, wq = w&3 selects columns.
// Wave pairs (rh=0/1, same wq) issue identical weight loads -> WGP$ hits,
// halving L2 weight traffic vs the 16-row version.
// Layer1 hid chunk (32x64) staged in LDS (bf16); layer2 accumulated in
// 12 persistent 16x16 f32 WMMA accumulators per wave. hid never hits HBM.
// ---------------------------------------------------------------------------
__global__ __launch_bounds__(256) void expert_ffn_kernel(
    const unsigned short* __restrict__ h,    // [B][D] bf16
    const unsigned short* __restrict__ w1t,  // [E][H][D] bf16
    const float* __restrict__ b1,            // [E][H]
    const unsigned short* __restrict__ w2t,  // [E][D][H] bf16
    const float* __restrict__ b2,            // [E][D]
    float* __restrict__ eo)                  // [B][E][D]
{
    constexpr int LDH = Dd + 8;  // 776 halfs: padded row stride -> conflict-free b128
    constexpr int LDC = 72;      // 64 + 8 pad
    __shared__ __attribute__((aligned(16))) unsigned short sh_h[32 * LDH];
    __shared__ __attribute__((aligned(16))) unsigned short sh_hid[32 * LDC];

    const int e    = blockIdx.y;
    const int row0 = blockIdx.x * 32;
    const int tid  = threadIdx.x;
    const int w    = tid >> 5;
    const int lane = tid & 31;
    const int lh   = lane & 15;
    const int hi   = lane >> 4;
    const int rh   = w >> 2;     // row-half: 0 or 1
    const int wq   = w & 3;      // column group: 0..3

    // Stage the 32x768 bf16 h-tile into LDS (async-to-LDS if available)
    for (int idx = tid; idx < 32 * (Dd / 8); idx += 256) {
        const int r = idx / (Dd / 8);
        const int c = (idx % (Dd / 8)) * 8;
        const unsigned short* gp = &h[(size_t)(row0 + r) * Dd + c];
        unsigned short* lp = &sh_h[r * LDH + c];
#if HAVE_ASYNC_LDS
        __builtin_amdgcn_global_load_async_to_lds_b128(
            (gv4i_p)(void*)const_cast<unsigned short*>(gp),
            (lv4i_p)(void*)lp, 0, 0);
#else
        *(v8u16*)lp = *(const v8u16*)gp;
#endif
    }
#if HAVE_ASYNC_LDS
#if __has_builtin(__builtin_amdgcn_s_wait_asynccnt)
    __builtin_amdgcn_s_wait_asynccnt(0);
#else
    asm volatile("s_wait_asynccnt 0" ::: "memory");
#endif
#endif
    __syncthreads();

    v8f acc[12];
#pragma unroll
    for (int t = 0; t < 12; ++t) acc[t] = (v8f){0.f, 0.f, 0.f, 0.f, 0.f, 0.f, 0.f, 0.f};

    const unsigned short* w1e = w1t + (size_t)e * Hh * Dd;
    const unsigned short* w2e = w2t + (size_t)e * Dd * Hh;

    for (int hc = 0; hc < Hh; hc += 64) {
        // ----- Layer 1: this wave computes hid tile (rows rh*16.., cols wq*16..) -----
        const int ncol = hc + wq * 16 + lh;
        const unsigned short* brow = w1e + (size_t)ncol * Dd;
        if (hc + 64 < Hh)  // prefetch next chunk's W1 row (global_prefetch_b8)
            __builtin_prefetch(w1e + (size_t)(ncol + 64) * Dd, 0, 3);

        v8f c1 = {0.f, 0.f, 0.f, 0.f, 0.f, 0.f, 0.f, 0.f};
        for (int k = 0; k < Dd; k += 32) {
            Frag16 a, b;
            a.h[0] = *(const v8u16*)&sh_h[(rh * 16 + lh) * LDH + k + hi * 8];
            a.h[1] = *(const v8u16*)&sh_h[(rh * 16 + lh) * LDH + k + 16 + hi * 8];
            b.h[0] = *(const v8u16*)&brow[k + hi * 8];
            b.h[1] = *(const v8u16*)&brow[k + 16 + hi * 8];
            c1 = __builtin_amdgcn_wmma_f32_16x16x32_bf16(false, a.v, false, b.v,
                                                         (short)0, c1, false, false);
        }
        const float bias1 = b1[(size_t)e * Hh + ncol];
        const int nl = wq * 16 + lh;
#pragma unroll
        for (int i = 0; i < 8; ++i) {
            sh_hid[(rh * 16 + i + hi * 8) * LDC + nl] = f2bfu(gelu_exact(c1[i] + bias1));
        }
        __syncthreads();

        // ----- Layer 2: accumulate hid chunk into 12 persistent tiles -----
#pragma unroll
        for (int kk = 0; kk < 64; kk += 32) {
            Frag16 a;
            a.h[0] = *(const v8u16*)&sh_hid[(rh * 16 + lh) * LDC + kk + hi * 8];
            a.h[1] = *(const v8u16*)&sh_hid[(rh * 16 + lh) * LDC + kk + 16 + hi * 8];
#pragma unroll
            for (int t = 0; t < 12; ++t) {
                const int n = (wq * 12 + t) * 16 + lh;
                const unsigned short* b2row = w2e + (size_t)n * Hh + hc + kk + hi * 8;
                Frag16 b;
                b.h[0] = *(const v8u16*)&b2row[0];
                b.h[1] = *(const v8u16*)&b2row[16];
                acc[t] = __builtin_amdgcn_wmma_f32_16x16x32_bf16(false, a.v, false, b.v,
                                                                 (short)0, acc[t], false, false);
            }
        }
        __syncthreads();  // protect sh_hid before next chunk overwrites it
    }

    // Epilogue: bias + store expert_outputs [B][E][D]
#pragma unroll
    for (int t = 0; t < 12; ++t) {
        const int n = (wq * 12 + t) * 16 + lh;
        const float bias2 = b2[(size_t)e * Dd + n];
#pragma unroll
        for (int i = 0; i < 8; ++i) {
            const int row = row0 + rh * 16 + i + hi * 8;
            eo[((size_t)row * Ee + e) * Dd + n] = acc[t][i] + bias2;
        }
    }
}

// ---------------------------------------------------------------------------
// Top-2 combine: result[b][d] = w0*eo[b][i0][d] + w1*eo[b][i1][d]
// ---------------------------------------------------------------------------
__global__ __launch_bounds__(256) void combine_kernel(const float* __restrict__ eo,
                                                      const float* __restrict__ cw,
                                                      float* __restrict__ result) {
    const size_t idx = (size_t)blockIdx.x * 256 + threadIdx.x;
    if (idx >= (size_t)Bn * Dd) return;
    const int row = (int)(idx / Dd);
    const int d   = (int)(idx % Dd);
    const float* c = cw + (size_t)row * 4;
    const int i0 = __float_as_int(c[2]);
    const int i1 = __float_as_int(c[3]);
    result[idx] = c[0] * eo[((size_t)row * Ee + i0) * Dd + d] +
                  c[1] * eo[((size_t)row * Ee + i1) * Dd + d];
}

// ---------------------------------------------------------------------------
extern "C" void kernel_launch(void* const* d_in, const int* in_sizes, int n_in,
                              void* d_out, int out_size, void* d_ws, size_t ws_size,
                              hipStream_t stream) {
    (void)in_sizes; (void)n_in; (void)out_size; (void)ws_size;
    const float* pooler = (const float*)d_in[0];
    const float* ln_g   = (const float*)d_in[1];
    const float* ln_b   = (const float*)d_in[2];
    const float* gw1    = (const float*)d_in[3];
    const float* gb1    = (const float*)d_in[4];
    const float* gw2    = (const float*)d_in[5];
    const float* gb2    = (const float*)d_in[6];
    const float* ew1    = (const float*)d_in[7];
    const float* eb1    = (const float*)d_in[8];
    const float* ew2    = (const float*)d_in[9];
    const float* eb2    = (const float*)d_in[10];

    float* out      = (float*)d_out;
    float* result   = out;                                  // [B][D]
    float* gate_out = out + (size_t)Bn * Dd;                // [B][E]
    float* eo       = gate_out + (size_t)Bn * Ee;           // [B][E][D]

    char* ws = (char*)d_ws;
    unsigned short* h_bf  = (unsigned short*)ws; ws += (size_t)Bn * Dd * 2;
    unsigned short* g_bf  = (unsigned short*)ws; ws += (size_t)Bn * Gg * 2;
    unsigned short* gw1t  = (unsigned short*)ws; ws += (size_t)Gg * Dd * 2;
    unsigned short* ew1t  = (unsigned short*)ws; ws += (size_t)Ee * Hh * Dd * 2;
    unsigned short* ew2t  = (unsigned short*)ws; ws += (size_t)Ee * Dd * Hh * 2;
    float*          cw    = (float*)ws;          ws += (size_t)Bn * 4 * 4;

    // Weight conversion / transpose
    cvt_gw1_kernel<<<(Gg * Dd + 255) / 256, 256, 0, stream>>>(gw1, gw1t);
    {
        const size_t tot = (size_t)Ee * Hh * Dd;
        cvt_ew1_kernel<<<(unsigned)((tot + 255) / 256), 256, 0, stream>>>(ew1, ew1t);
        cvt_ew2_kernel<<<(unsigned)((tot + 255) / 256), 256, 0, stream>>>(ew2, ew2t);
    }
    // LayerNorm
    ln_kernel<<<Bn, 256, 0, stream>>>(pooler, ln_g, ln_b, h_bf);
    // Gating
    gate1_kernel<<<dim3(Bn / 16, Gg / 16), 32, 0, stream>>>(h_bf, gw1t, gb1, g_bf);
    gate2_kernel<<<(Bn + 255) / 256, 256, 0, stream>>>(g_bf, gw2, gb2, gate_out, cw);
    // Dense expert stack (fused two-layer FFN, WMMA bf16, 32-row tiles)
    expert_ffn_kernel<<<dim3(Bn / 32, Ee), 256, 0, stream>>>(h_bf, ew1t, eb1, ew2t, eb2, eo);
    // Top-2 combine
    combine_kernel<<<(unsigned)(((size_t)Bn * Dd + 255) / 256), 256, 0, stream>>>(eo, cw, result);
}
